// CostVolumeLayer_54786602828104
// MI455X (gfx1250) — matrix operand order repaired and checked by reading, not compiled, not run on hardware
//
#include <hip/hip_runtime.h>

// CDNA5 / gfx1250 cost-volume kernel: banded-matmul via v_wmma_f32_16x16x32_bf16
// with bf16 hi/lo split (3 WMMAs per K-chunk) for fp32-class accuracy.
// Boundary handling: clamped unconditional loads + zeroing at store time
// (a lane's B column only affects that same lane's output column).

typedef __attribute__((ext_vector_type(16))) __bf16 v16bf;
typedef __attribute__((ext_vector_type(8)))  float  v8f;

#define NB  4
#define NC  128
#define NH  96
#define NW  160
#define ND  81
#define NXT 10   // NW / 16 x-tiles per row

// Channel index for displacement (dy,dx), matching the reference's
// _displacements() ordering: (0,0); then per ring i=1..4: (-i,0),(i,0),(0,-i),(0,i),
// then per j=1..4 the quadrants (-i,-j),(i,j),(-i,j),(i,-j).
__device__ __forceinline__ int disp_index(int dy, int dx) {
  int ady = dy < 0 ? -dy : dy;
  int adx = dx < 0 ? -dx : dx;
  if ((ady | adx) == 0) return 0;
  if (adx == 0) return 1 + (ady - 1) * 20 + (dy > 0 ? 1 : 0);
  if (ady == 0) return 3 + (adx - 1) * 20 + (dx > 0 ? 1 : 0);
  int quad = (dy < 0) ? (dx < 0 ? 0 : 2) : (dx > 0 ? 1 : 3);
  return 5 + (ady - 1) * 20 + (adx - 1) * 4 + quad;
}

__launch_bounds__(288)
__global__ void costvol_wmma_bf16x3(const float* __restrict__ src,
                                    const float* __restrict__ tgt,
                                    float* __restrict__ out) {
  // src tile staged transposed [x][c] with pad to dodge LDS bank conflicts
  __shared__ float sS[16][NC + 4];

  const int bid = blockIdx.x;
  const int b   = bid / (NH * NXT);
  const int r0  = bid % (NH * NXT);
  const int y   = r0 / NXT;
  const int x0  = (r0 % NXT) * 16;

  const int tid = threadIdx.x;

  // ---- cooperative stage: src[b, 0..127, y, x0..x0+15] -> LDS (coalesced reads) ----
  for (int idx = tid; idx < NC * 16; idx += 288) {
    int c = idx >> 4, xx = idx & 15;
    sS[xx][c] = src[(((size_t)b * NC + c) * NH + y) * NW + x0 + xx];
  }
  __syncthreads();

  const int  wave  = tid >> 5;        // 0..8, one wave per dy
  const int  lane  = tid & 31;
  const int  dy    = wave - 4;
  const int  yy    = y + dy;
  const bool yok   = (yy >= 0) && (yy < NH);   // uniform per wave
  const int  n     = lane & 15;       // A: M index; B: N index
  const int  hi16  = lane >> 4;       // lane half
  const int  koffA = hi16 * 8;        // A K-offset per ISA 16-bit A layout
  const int  koffB = hi16 * 8;        // B K-offset, same layout as A

  // ---- build A operands (src column x0+M, all 128 channels), bf16 hi/lo split ----
  v16bf Ahi[4], Alo[4];
#pragma unroll
  for (int kc = 0; kc < 4; ++kc) {
#pragma unroll
    for (int j = 0; j < 8; ++j) {
      // VGPR j holds K = kA, kA+1 : kA = 32*kc + 2*(j&3) + 16*(j>>2) + koffA
      int k = kc * 32 + (j & 3) * 2 + (j >> 2) * 16 + koffA;
      float f0 = sS[n][k];
      float f1 = sS[n][k + 1];
      __bf16 h0 = (__bf16)f0, h1 = (__bf16)f1;
      Ahi[kc][2 * j]     = h0;
      Ahi[kc][2 * j + 1] = h1;
      Alo[kc][2 * j]     = (__bf16)(f0 - (float)h0);
      Alo[kc][2 * j + 1] = (__bf16)(f1 - (float)h1);
    }
  }

  const size_t cstride = (size_t)NH * NW;
  const int    yc      = yy < 0 ? 0 : (yy >= NH ? NH - 1 : yy);  // clamped row
  const float* trow    = tgt + ((size_t)b * NC * NH + (size_t)yc) * NW;

  // ---- two B n-tiles cover the full dx band [-4,4] of the correlation matrix ----
#pragma unroll
  for (int t = 0; t < 2; ++t) {
    const int  xcol = x0 - 4 + 16 * t + n;            // tgt column for this lane
    const int  xc   = xcol < 0 ? 0 : (xcol >= NW ? NW - 1 : xcol);
    const bool ok   = yok && (xcol >= 0) && (xcol < NW);   // zero-pad predicate
    const float* pB = trow + xc;                      // per-lane base; k via imm offsets

    v8f acc = {};
    if (yok) {   // uniform scalar branch: EXEC stays all-ones around WMMA
#pragma unroll
      for (int kc = 0; kc < 4; ++kc) {
        v16bf bhi, blo;
#pragma unroll
        for (int j = 0; j < 8; ++j) {
          int k = kc * 32 + (j & 3) * 2 + (j >> 2) * 16 + koffB;  // VGPR j holds K = k, k+1
          float g0 = pB[(size_t)k * cstride];         // unconditional clamped loads
          float g1 = pB[(size_t)(k + 1) * cstride];
          __bf16 h0 = (__bf16)g0, h1 = (__bf16)g1;
          bhi[2 * j]     = h0;
          bhi[2 * j + 1] = h1;
          blo[2 * j]     = (__bf16)(g0 - (float)h0);
          blo[2 * j + 1] = (__bf16)(g1 - (float)h1);
        }
        // bf16x3: hi*hi + hi*lo + lo*hi accumulated in f32
        acc = __builtin_amdgcn_wmma_f32_16x16x32_bf16(false, Ahi[kc], false, bhi,
                                                      (short)0, acc, false, false);
        acc = __builtin_amdgcn_wmma_f32_16x16x32_bf16(false, Ahi[kc], false, blo,
                                                      (short)0, acc, false, false);
        acc = __builtin_amdgcn_wmma_f32_16x16x32_bf16(false, Alo[kc], false, bhi,
                                                      (short)0, acc, false, false);
      }
    }

    // ---- scatter in-band elements: C/D layout M = j + 8*hi16, N = lane&15 ----
    // Output element (M, N=n) has x+dx == this lane's xcol, so `ok` is exactly
    // the zero-padding predicate for every element this lane stores.
#pragma unroll
    for (int j = 0; j < 8; ++j) {
      int M  = j + hi16 * 8;
      int dx = n - M - 4 + 16 * t;
      if (dx >= -4 && dx <= 4) {
        int   d = disp_index(dy, dx);
        float v = ok ? acc[j] * (1.0f / 81.0f) : 0.0f;
        out[(((size_t)b * ND + d) * NH + y) * NW + x0 + M] = v;
      }
    }
  }
}

extern "C" void kernel_launch(void* const* d_in, const int* in_sizes, int n_in,
                              void* d_out, int out_size, void* d_ws, size_t ws_size,
                              hipStream_t stream) {
  const float* src = (const float*)d_in[0];
  const float* tgt = (const float*)d_in[1];
  float* out = (float*)d_out;
  (void)in_sizes; (void)n_in; (void)out_size; (void)d_ws; (void)ws_size;

  dim3 grid(NB * NH * NXT);   // 4 * 96 * 10 = 3840 workgroups
  dim3 block(288);            // 9 wave32s, one per dy
  hipLaunchKernelGGL(costvol_wmma_bf16x3, grid, block, 0, stream, src, tgt, out);
}
